// DenseFp8_79620103733323
// MI455X (gfx1250) — 8-bit matrix operand order repaired and checked by reading, compiled, not run on hardware
//
#include <hip/hip_runtime.h>
#include <hip/hip_bf16.h>


typedef int   v16i __attribute__((ext_vector_type(16)));
typedef float v8f  __attribute__((ext_vector_type(8)));
typedef int   v4i  __attribute__((ext_vector_type(4)));

#define E4M3_MAX 448.0f

#if __has_builtin(__builtin_amdgcn_global_load_async_to_lds_b128)
#define HAVE_ASYNC_LDS 1
#else
#define HAVE_ASYNC_LDS 0
#endif

typedef __attribute__((address_space(1))) v4i glb_v4i;
typedef __attribute__((address_space(3))) v4i lds_v4i;

__device__ inline glb_v4i* glb4(const void* p) {
    return (glb_v4i*)(uintptr_t)p;
}
__device__ inline lds_v4i* lds4(void* p) {
    return (lds_v4i*)(unsigned)(uintptr_t)p;
}

__device__ inline void wait_async0() {
#if __has_builtin(__builtin_amdgcn_s_wait_asynccnt)
    __builtin_amdgcn_s_wait_asynccnt(0);
#else
    asm volatile("s_wait_asynccnt 0x0" ::: "memory");
#endif
}

__device__ inline unsigned pack4_e4m3(float c0, float c1, float c2, float c3) {
#if __has_builtin(__builtin_amdgcn_cvt_pk_fp8_f32)
    int w = __builtin_amdgcn_cvt_pk_fp8_f32(c0, c1, 0, false);
    w = __builtin_amdgcn_cvt_pk_fp8_f32(c2, c3, w, true);
    return (unsigned)w;
#else
    auto enc = [](float f) -> unsigned {
        unsigned u = __float_as_uint(f);
        unsigned sign = (u >> 31) << 7;
        float a = fabsf(f);
        if (a < 1.0f / 1024.0f) {
            int m = (int)(a * 512.0f + 0.5f);
            if (m >= 8) return sign | 0x08;
            return sign | (unsigned)m;
        }
        int   exp = (int)((u >> 23) & 0xff) - 127;
        unsigned man = u & 0x7fffff;
        unsigned m3 = (man + 0x0FFFFF + ((man >> 20) & 1)) >> 20;
        if (m3 >= 8) { m3 = 0; exp++; }
        if (exp > 8) { exp = 8; m3 = 6; }
        if (exp < -6) {
            int m = (int)(a * 512.0f + 0.5f);
            if (m >= 8) return sign | 0x08;
            return sign | (unsigned)m;
        }
        return sign | ((unsigned)(exp + 7) << 3) | m3;
    };
    return enc(c0) | (enc(c1) << 8) | (enc(c2) << 16) | (enc(c3) << 24);
#endif
}

__device__ inline float block_max256(float v) {
    __shared__ float red[256];
    int t = threadIdx.x;
    red[t] = v;
    __syncthreads();
    #pragma unroll
    for (int s = 128; s > 0; s >>= 1) {
        if (t < s) red[t] = fmaxf(red[t], red[t + s]);
        __syncthreads();
    }
    return red[0];
}

__global__ __launch_bounds__(256)
void k_quant_x(const float* __restrict__ x, unsigned* __restrict__ qx,
               const float* __restrict__ scale_p, float* __restrict__ amax,
               long long n4) {
    const float inv = 1.0f / scale_p[0];
    float m = 0.0f;
    long long i = (long long)blockIdx.x * blockDim.x + threadIdx.x;
    long long stride = (long long)gridDim.x * blockDim.x;
    const float4* x4 = (const float4*)x;
    for (; i < n4; i += stride) {
        float4 v = x4[i];
        m = fmaxf(m, fmaxf(fmaxf(fabsf(v.x), fabsf(v.y)),
                           fmaxf(fabsf(v.z), fabsf(v.w))));
        float c0 = fminf(fmaxf(v.x * inv, -E4M3_MAX), E4M3_MAX);
        float c1 = fminf(fmaxf(v.y * inv, -E4M3_MAX), E4M3_MAX);
        float c2 = fminf(fmaxf(v.z * inv, -E4M3_MAX), E4M3_MAX);
        float c3 = fminf(fmaxf(v.w * inv, -E4M3_MAX), E4M3_MAX);
        qx[i] = pack4_e4m3(c0, c1, c2, c3);
    }
    float bm = block_max256(m);
    if (threadIdx.x == 0)
        atomicMax((unsigned*)amax, __float_as_uint(bm));
}

__global__ __launch_bounds__(256)
void k_quant_w_T(const float* __restrict__ W, unsigned* __restrict__ qkT,
                 const float* __restrict__ scale_p, float* __restrict__ amax,
                 int D, int U) {
    __shared__ unsigned ldsq[64 * 17];
    const float inv = 1.0f / scale_p[0];
    const int u0 = blockIdx.x * 64;
    const int d0 = blockIdx.y * 64;
    const int t = threadIdx.x;
    const int r = t >> 4;
    const int c4 = t & 15;

    float m = 0.0f;
    #pragma unroll
    for (int rr = 0; rr < 64; rr += 16) {
        const int d = d0 + r + rr;
        float4 v = *(const float4*)(W + (long long)d * U + u0 + c4 * 4);
        m = fmaxf(m, fmaxf(fmaxf(fabsf(v.x), fabsf(v.y)),
                           fmaxf(fabsf(v.z), fabsf(v.w))));
        float c0 = fminf(fmaxf(v.x * inv, -E4M3_MAX), E4M3_MAX);
        float c1 = fminf(fmaxf(v.y * inv, -E4M3_MAX), E4M3_MAX);
        float c2 = fminf(fmaxf(v.z * inv, -E4M3_MAX), E4M3_MAX);
        float c3 = fminf(fmaxf(v.w * inv, -E4M3_MAX), E4M3_MAX);
        ldsq[(r + rr) * 17 + c4] = pack4_e4m3(c0, c1, c2, c3);
    }
    __syncthreads();
    #pragma unroll
    for (int rr = 0; rr < 64; rr += 16) {
        const int ul = r + rr;
        unsigned w = 0;
        #pragma unroll
        for (int j = 0; j < 4; ++j) {
            const int dl = c4 * 4 + j;
            unsigned byte = (ldsq[dl * 17 + (ul >> 2)] >> ((ul & 3) * 8)) & 0xffu;
            w |= byte << (8 * j);
        }
        qkT[((long long)(u0 + ul) * D + d0) / 4 + c4] = w;
    }
    __syncthreads();
    float bm = block_max256(m);
    if (threadIdx.x == 0)
        atomicMax((unsigned*)amax, __float_as_uint(bm));
}

#define TILE_BYTES   16384
#define BUF_BYTES    32768

__device__ __forceinline__ void compute_tile(const char* ldsc, unsigned bufbase,
                                             int wm, int wn, int lgrp, int lmod,
                                             v8f acc[4][2]) {
    v16i bfrag[2];
    #pragma unroll
    for (int ni = 0; ni < 2; ++ni) {
        const int col = wn * 32 + ni * 16 + lmod;
        const unsigned swz = (unsigned)((col >> 1) & 7);
        const unsigned base = bufbase + TILE_BYTES + (unsigned)col * 128u;
        #pragma unroll
        for (int q = 0; q < 4; ++q) {
            const unsigned c = (unsigned)(q * 2 + lgrp);
            uint4 d = *(const uint4*)(ldsc + base + ((c ^ swz) << 4));
            bfrag[ni][4 * q + 0] = (int)d.x;
            bfrag[ni][4 * q + 1] = (int)d.y;
            bfrag[ni][4 * q + 2] = (int)d.z;
            bfrag[ni][4 * q + 3] = (int)d.w;
        }
    }
    #pragma unroll
    for (int mi = 0; mi < 4; ++mi) {
        const int row = wm * 64 + mi * 16 + lmod;
        const unsigned swz = (unsigned)((row >> 1) & 7);
        const unsigned base = bufbase + (unsigned)row * 128u;
        v16i afrag;
        #pragma unroll
        for (int h = 0; h < 2; ++h)
            #pragma unroll
            for (int p = 0; p < 4; ++p) {
                const unsigned c = (unsigned)(h * 4 + (p >> 1) * 2 + lgrp);
                uint2 d = *(const uint2*)(ldsc + base + ((c ^ swz) << 4) + (unsigned)((p & 1) * 8));
                afrag[h * 8 + 2 * p + 0] = (int)d.x;
                afrag[h * 8 + 2 * p + 1] = (int)d.y;
            }
        #pragma unroll
        for (int ni = 0; ni < 2; ++ni)
            acc[mi][ni] = __builtin_amdgcn_wmma_f32_16x16x128_fp8_fp8(
                afrag, bfrag[ni], (short)0, acc[mi][ni], false, false);
    }
}

__global__ __launch_bounds__(256)
void k_gemm_fp8(const unsigned char* __restrict__ qx,
                const unsigned char* __restrict__ qkT,
                const float* __restrict__ bias,
                const float* __restrict__ in_scale,
                const float* __restrict__ k_scale,
                float* __restrict__ out,
                int K, int U) {
    __shared__ uint4 smem[2 * BUF_BYTES / 16];
    char* ldsc = (char*)smem;

    const int tid  = threadIdx.x;
    const int lane = tid & 31;
    const int wave = tid >> 5;
    const int wm   = wave >> 2;
    const int wn   = wave & 3;
    const int lgrp = lane >> 4;
    const int lmod = lane & 15;

    const int blockM = blockIdx.y * 128;
    const int blockN = blockIdx.x * 128;

    const int cr = tid >> 3;
    const int cc = tid & 7;
    const unsigned char* aSrc = qx  + (long long)(blockM + cr) * K + cc * 16;
    const unsigned char* bSrc = qkT + (long long)(blockN + cr) * K + cc * 16;

    v8f zero = {0, 0, 0, 0, 0, 0, 0, 0};
    v8f acc[4][2];
    #pragma unroll
    for (int mi = 0; mi < 4; ++mi)
        #pragma unroll
        for (int ni = 0; ni < 2; ++ni) acc[mi][ni] = zero;

#if HAVE_ASYNC_LDS
    auto issue = [&](unsigned bufbase, int k0) {
        #pragma unroll
        for (int i = 0; i < 4; ++i) {
            const int r = cr + i * 32;
            const unsigned sw = (unsigned)((r >> 1) & 7);
            const unsigned dst = bufbase + (unsigned)r * 128u + (((unsigned)cc ^ sw) << 4);
            __builtin_amdgcn_global_load_async_to_lds_b128(
                glb4(aSrc + (long long)(i * 32) * K + k0), lds4(ldsc + dst), 0, 0);
            __builtin_amdgcn_global_load_async_to_lds_b128(
                glb4(bSrc + (long long)(i * 32) * K + k0),
                lds4(ldsc + dst + TILE_BYTES), 0, 0);
        }
    };

    issue(0, 0);
    wait_async0();
    __syncthreads();

    for (int k0 = 0; k0 < K; k0 += 128) {
        const unsigned buf = ((unsigned)(k0 >> 7) & 1u) * BUF_BYTES;
        const bool more = (k0 + 128 < K);
        if (more) issue(buf ^ BUF_BYTES, k0 + 128);
        compute_tile(ldsc, buf, wm, wn, lgrp, lmod, acc);
        if (more) {
            wait_async0();
            __syncthreads();
        }
    }
#else
    for (int k0 = 0; k0 < K; k0 += 128) {
        __syncthreads();
        #pragma unroll
        for (int i = 0; i < 4; ++i) {
            uint4 va = *(const uint4*)(aSrc + (long long)(i * 32) * K + k0);
            uint4 vb = *(const uint4*)(bSrc + (long long)(i * 32) * K + k0);
            const int r = cr + i * 32;
            const unsigned sw = (unsigned)((r >> 1) & 7);
            const unsigned dst = (unsigned)r * 128u + (((unsigned)cc ^ sw) << 4);
            *(uint4*)(ldsc + dst) = va;
            *(uint4*)(ldsc + dst + TILE_BYTES) = vb;
        }
        __syncthreads();
        if (k0 + 128 < K && (cc & 3) == 0) {
            #pragma unroll
            for (int i = 0; i < 4; ++i) {
                __builtin_prefetch(aSrc + (long long)(i * 32) * K + k0 + 128, 0, 0);
                __builtin_prefetch(bSrc + (long long)(i * 32) * K + k0 + 128, 0, 0);
            }
        }
        compute_tile(ldsc, 0, wm, wn, lgrp, lmod, acc);
    }
#endif

    const float s = in_scale[0] * k_scale[0];
    #pragma unroll
    for (int mi = 0; mi < 4; ++mi) {
        const int row0 = blockM + wm * 64 + mi * 16 + lgrp * 8;
        #pragma unroll
        for (int ni = 0; ni < 2; ++ni) {
            const int col = blockN + wn * 32 + ni * 16 + lmod;
            const float b = bias[col];
            float* po = out + (long long)row0 * U + col;
            #pragma unroll
            for (int r = 0; r < 8; ++r) {
                *po = acc[mi][ni][r] * s + b;
                po += U;
            }
        }
    }
}

__global__ void k_finalize(const float* __restrict__ amax2,
                           const float* __restrict__ in_hist,
                           const float* __restrict__ k_hist,
                           float* __restrict__ tail) {
    if (threadIdx.x == 0 && blockIdx.x == 0) {
        float mx = amax2[0];
        tail[2] = amax2[0];
        for (int i = 1; i < 16; ++i) { float h = in_hist[i - 1]; tail[2 + i] = h; mx = fmaxf(mx, h); }
        mx = fmaxf(mx, 0.0009765625f);
        tail[0] = 1.1f * mx / E4M3_MAX;

        mx = amax2[1];
        tail[18] = amax2[1];
        for (int i = 1; i < 16; ++i) { float h = k_hist[i - 1]; tail[18 + i] = h; mx = fmaxf(mx, h); }
        mx = fmaxf(mx, 0.0009765625f);
        tail[1] = 1.1f * mx / E4M3_MAX;
    }
}

extern "C" void kernel_launch(void* const* d_in, const int* in_sizes, int n_in,
                              void* d_out, int out_size, void* d_ws, size_t ws_size,
                              hipStream_t stream) {
    const float* x        = (const float*)d_in[0];
    const float* W        = (const float*)d_in[1];
    const float* bias     = (const float*)d_in[2];
    const float* in_scale = (const float*)d_in[3];
    const float* k_scale  = (const float*)d_in[4];
    const float* in_hist  = (const float*)d_in[7];
    const float* k_hist   = (const float*)d_in[8];

    const int N = 8192, D = 4096, U = 4096;

    unsigned char* qx  = (unsigned char*)d_ws;
    unsigned char* qkT = qx + (long long)N * D;
    float*         amax = (float*)(qkT + (long long)D * U);

    float* out  = (float*)d_out;
    float* tail = out + (long long)N * U;

    (void)hipMemsetAsync(amax, 0, 2 * sizeof(float), stream);

    k_quant_x<<<4096, 256, 0, stream>>>(x, (unsigned*)qx, in_scale, amax + 0,
                                        (long long)N * D / 4);
    k_quant_w_T<<<dim3(U / 64, D / 64), 256, 0, stream>>>(W, (unsigned*)qkT, k_scale,
                                                          amax + 1, D, U);
    k_gemm_fp8<<<dim3(U / 128, N / 128), 256, 0, stream>>>(qx, qkT, bias, in_scale,
                                                           k_scale, out, D, U);
    k_finalize<<<1, 32, 0, stream>>>(amax, in_hist, k_hist, tail);
}
